// DeepseekV3Moe_71657234367032
// MI455X (gfx1250) — matrix operand order repaired and checked by reading, not compiled, not run on hardware
//
#include <hip/hip_runtime.h>

// ---------------- problem constants ----------------
constexpr int T = 1024, H = 2048, II = 1408, E = 32;
constexpr int TOPK = 6, NGROUP = 4, GSIZE = E / NGROUP;
constexpr float ROUTED_SCALE = 2.5f;
constexpr int BLK = 256;          // 8 waves of 32
constexpr int MT  = 16;           // token tile (one WMMA M tile)
constexpr int NT1 = II / 16;      // 88 N tiles in gate/up GEMM
constexpr int NT2 = H  / 16;      // 128 N tiles in down GEMM
constexpr int CAP = 1024;         // per-expert slot capacity

// ---------------- vector types ----------------
typedef __attribute__((ext_vector_type(16))) __bf16 v16bf;
typedef __attribute__((ext_vector_type(8)))  float  v8f;
typedef __attribute__((ext_vector_type(4)))  unsigned int v4u;
typedef __attribute__((ext_vector_type(8)))  int v8i;
typedef __attribute__((ext_vector_type(4)))  int v4i;

union V16 { v16bf bf; uint4 q[2]; };

__device__ inline v8f wmma_bf16(v16bf a, v16bf b, v8f c) {
  return __builtin_amdgcn_wmma_f32_16x16x32_bf16(false, a, false, b, (short)0, c, false, false);
}

// fast sigmoid/silu helpers: v_rcp_f32 instead of IEEE division expansion
__device__ inline float fast_sigmoid(float v) {
  return __builtin_amdgcn_rcpf(1.f + __expf(-v));
}
__device__ inline float fast_silu(float g) {
  return g * __builtin_amdgcn_rcpf(1.f + __expf(-g));
}

// A fragment (16x32 bf16, row-major LDS, stride lda), K base k0 (contiguous k = 16*half + s).
__device__ inline v16bf lds_load_a(const __bf16* s, int lda, int k0) {
  int lane = threadIdx.x & 31;
  int r = lane & 15;
  int hf = lane >> 4;
  const __bf16* p = s + r * lda + k0 + hf * 16;
  V16 u;
  u.q[0] = *(const uint4*)p;
  u.q[1] = *(const uint4*)(p + 8);
  return u.bf;
}

// B fragment (32x16) from global fp32 at (k0,n0), row stride ldb.
__device__ inline v16bf glb_load_b(const float* B, size_t ldb) {
  int lane = threadIdx.x & 31;
  int r = lane & 15;
  int hf = lane >> 4;
  const float* p = B + (size_t)(hf * 16) * ldb + r;
  V16 u;
#pragma unroll
  for (int kk = 0; kk < 16; ++kk) u.bf[kk] = (__bf16)p[(size_t)kk * ldb];
  return u.bf;
}

// ---------------- router: 1 wave per token, 1 lane per expert ----------------
__global__ __launch_bounds__(32) void router_kernel(
    const float* __restrict__ x, const float* __restrict__ gw,
    const float* __restrict__ ebias, float* __restrict__ topk_w,
    int* __restrict__ topk_id) {
  __shared__ float xsh[H];
  __shared__ float s_sc[E], s_sfc[E], s_g[NGROUP];
  int t = blockIdx.x, lane = threadIdx.x;
  for (int i = lane; i < H / 4; i += 32)
    ((float4*)xsh)[i] = ((const float4*)(x + (size_t)t * H))[i];
  __syncthreads();
  float acc = 0.f;
  const float* w = gw + (size_t)lane * H;
  for (int k = 0; k < H; k += 4) {
    float4 a = *(const float4*)(xsh + k);
    float4 b = *(const float4*)(w + k);
    acc += a.x * b.x + a.y * b.y + a.z * b.z + a.w * b.w;
  }
  float sc = fast_sigmoid(acc);
  s_sc[lane] = sc;
  s_sfc[lane] = sc + ebias[lane];               // bias only for selection
  __syncthreads();
  if (lane < NGROUP) {                          // per-group top-2 sum
    float m1 = -1e30f, m2 = -1e30f;
    for (int i = 0; i < GSIZE; ++i) {
      float v = s_sfc[lane * GSIZE + i];
      if (v > m1) { m2 = m1; m1 = v; } else if (v > m2) m2 = v;
    }
    s_g[lane] = m1 + m2;
  }
  __syncthreads();
  if (lane == 0) {
    int g1 = 0;
    for (int g = 1; g < NGROUP; ++g) if (s_g[g] > s_g[g1]) g1 = g;
    int g2 = -1;
    for (int g = 0; g < NGROUP; ++g) { if (g == g1) continue; if (g2 < 0 || s_g[g] > s_g[g2]) g2 = g; }
    float masked[E];
    for (int e = 0; e < E; ++e) {
      int g = e / GSIZE;
      masked[e] = (g == g1 || g == g2) ? s_sfc[e] : -1e30f;
    }
    float sum = 0.f; int ids[TOPK]; float ws[TOPK];
    for (int k = 0; k < TOPK; ++k) {
      int best = 0;
      for (int e = 1; e < E; ++e) if (masked[e] > masked[best]) best = e;
      ids[k] = best; ws[k] = s_sc[best]; sum += ws[k];
      masked[best] = -2e30f;
    }
    float inv = __builtin_amdgcn_rcpf(sum + 1e-20f);
    for (int k = 0; k < TOPK; ++k) {
      topk_id[t * TOPK + k] = ids[k];
      topk_w[t * TOPK + k] = ws[k] * inv;
    }
  }
}

// ---------------- dispatch ----------------
__global__ void zero_counts(int* cnt) { if (threadIdx.x < E) cnt[threadIdx.x] = 0; }

__global__ __launch_bounds__(256) void scatter_kernel(
    const int* __restrict__ topk_id, const float* __restrict__ topk_w,
    int* __restrict__ cnt, int* __restrict__ slot_tok, float* __restrict__ slot_w) {
  int t = blockIdx.x * 256 + threadIdx.x;
  if (t >= T) return;
  for (int k = 0; k < TOPK; ++k) {
    int e = topk_id[t * TOPK + k];
    int pos = atomicAdd(&cnt[e], 1);
    slot_tok[e * CAP + pos] = t;
    slot_w[e * CAP + pos] = topk_w[t * TOPK + k];
  }
}

// ---------------- fused SwiGLU FFN for one 16-token tile ----------------
// LDS: fp32 x stage (TDM dest, 128KB) + bf16 x tile (64KB) + bf16 h tile (44KB).
// CDNA5 has 320KB LDS per WGP; this kernel is HBM-bound so 1 WG/WGP is fine.
template <bool ROUTED>
__global__ __launch_bounds__(BLK) void ffn_kernel(
    const float* __restrict__ x,     // [T,H]
    const float* __restrict__ wg,    // ROUTED: [E,H,I] else [H,I]
    const float* __restrict__ wu,
    const float* __restrict__ wd,    // ROUTED: [E,I,H] else [I,H]
    const int*   __restrict__ slot_tok,
    const float* __restrict__ slot_w,
    const int*   __restrict__ counts,
    float* __restrict__ out) {       // [T,H]
  extern __shared__ unsigned char smem[];
  float*  xstage = (float*)smem;                       // MT x H fp32 (TDM gather dest, LDS off 0)
  __bf16* xs = (__bf16*)(smem + (size_t)MT * H * 4);   // MT x H bf16
  __bf16* hs = xs + MT * H;                            // MT x II bf16
  int*    tok_s = (int*)(hs + MT * II);                // MT
  float*  scl_s = (float*)(tok_s + MT);                // MT

  const int e = ROUTED ? blockIdx.y : 0;
  const int tile = blockIdx.x;
  int valid = MT;
  if (ROUTED) {
    int cnt = counts[e];
    if (tile * MT >= cnt) return;                      // uniform early-exit
    int rem = cnt - tile * MT;
    valid = rem < MT ? rem : MT;
  }
  const int tid = threadIdx.x;
  const int wave = tid >> 5;
  const int lane = tid & 31;
  const int r = lane & 15;
  const int hf = lane >> 4;

  if (tid < MT) {
    int t; float w;
    if (ROUTED) {
      if (tid < valid) {
        t = slot_tok[e * CAP + tile * MT + tid];
        w = slot_w[e * CAP + tile * MT + tid] * ROUTED_SCALE;
      } else { t = -1; w = 0.f; }
    } else { t = tile * MT + tid; w = 1.f; }
    tok_s[tid] = t; scl_s[tid] = w;
  }

#if __has_builtin(__builtin_amdgcn_tensor_load_to_lds)
  // ---- TDM gather: DMA `valid` token rows of x (fp32) into LDS (ISA ch.8 D#) ----
  {
    unsigned long long ga = (unsigned long long)(const void*)x;
    v4u g0;
    g0[0] = 1u | (1u << 31);                 // count=1, gather_mode=1, 16-bit indices
    g0[1] = 0u;                              // lds_addr: xstage at dynamic-LDS offset 0
    g0[2] = (unsigned)(ga & 0xffffffffu);    // global_addr[31:0]
    g0[3] = (unsigned)((ga >> 32) & 0x01ffffffu) | (2u << 30);   // addr[56:32] | type=2
    v8i g1;
    g1[0] = (int)(2u << 16);                                     // data_size=4B
    g1[1] = (int)(((unsigned)H & 0xffffu) << 16);                // tensor_dim0 lo16
    g1[2] = (int)(((unsigned)H >> 16) & 0xffffu) | (int)(((unsigned)T & 0xffffu) << 16);
    g1[3] = (int)(((unsigned)T >> 16) & 0xffffu) | (int)(((unsigned)H & 0xffffu) << 16); // tile_dim0
    g1[4] = valid & 0xffff;                                      // tile_dim1 = #indices
    g1[5] = H;                                                   // tensor_dim0_stride lo32
    g1[6] = 0; g1[7] = 0;
    int idx[MT];
#pragma unroll
    for (int i = 0; i < MT; ++i) {
      int t = 0;
      if (i < valid) t = ROUTED ? slot_tok[e * CAP + tile * MT + i] : (tile * MT + i);
      idx[i] = __builtin_amdgcn_readfirstlane(t) & 0xffff;       // uniform row indices
    }
    v4i g2, g3;
#pragma unroll
    for (int j = 0; j < 4; ++j) {
      g2[j] = idx[2 * j]     | (idx[2 * j + 1] << 16);
      g3[j] = idx[8 + 2 * j] | (idx[9 + 2 * j] << 16);
    }
    v8i gpad = {0, 0, 0, 0, 0, 0, 0, 0};     // unused trailing group (6-arg builtin form)
    if (wave == 0) {                         // one wave issues the DMA
      __builtin_amdgcn_tensor_load_to_lds(g0, g1, g2, g3, gpad, 0);
      __builtin_amdgcn_s_wait_tensorcnt(0);
    }
  }
  __syncthreads();                            // publish TDM result + tok_s to all waves
  // one-time fp32 -> bf16 conversion into the WMMA A staging tile (zero-pad short tiles)
  for (int i = tid; i < MT * (H / 4); i += BLK) {
    int m = i >> 9, c4 = i & (H / 4 - 1);
    float4 v = make_float4(0.f, 0.f, 0.f, 0.f);
    if (m < valid) v = ((const float4*)(xstage + (size_t)m * H))[c4];
    __bf16* dst = xs + m * H + c4 * 4;
    dst[0] = (__bf16)v.x; dst[1] = (__bf16)v.y; dst[2] = (__bf16)v.z; dst[3] = (__bf16)v.w;
  }
#else
  // ---- fallback: per-thread global gather of the x tile ----
  __syncthreads();
  for (int i = tid; i < MT * (H / 4); i += BLK) {
    int m = i >> 9, c4 = i & (H / 4 - 1);
    int t = tok_s[m];
    float4 v = make_float4(0.f, 0.f, 0.f, 0.f);
    if (t >= 0) v = ((const float4*)(x + (size_t)t * H))[c4];
    __bf16* dst = xs + m * H + c4 * 4;
    dst[0] = (__bf16)v.x; dst[1] = (__bf16)v.y; dst[2] = (__bf16)v.z; dst[3] = (__bf16)v.w;
  }
#endif
  __syncthreads();

  const size_t eoff_gu = ROUTED ? (size_t)e * H * II : 0;
  const size_t eoff_d  = ROUTED ? (size_t)e * II * H : 0;
  const float* wge = wg + eoff_gu;
  const float* wue = wu + eoff_gu;
  const float* wde = wd + eoff_d;

  // ---- phase 1: g = x*Wg, u = x*Wu, h = silu(g)*u -> LDS (bf16) ----
  for (int nt = wave; nt < NT1; nt += BLK / 32) {
    const int n0 = nt * 16;
    v8f cg = {0, 0, 0, 0, 0, 0, 0, 0};
    v8f cu = {0, 0, 0, 0, 0, 0, 0, 0};
    for (int k0 = 0; k0 < H; k0 += 32) {
      // unconditional speculative prefetch (dropped if OOB) -> pulls next tile into GL2
      __builtin_prefetch(wge + (size_t)(k0 + 32) * II + n0, 0, 0);
      v16bf a  = lds_load_a(xs, H, k0);
      v16bf bg = glb_load_b(wge + (size_t)k0 * II + n0, II);
      v16bf bu = glb_load_b(wue + (size_t)k0 * II + n0, II);
      cg = wmma_bf16(a, bg, cg);
      cu = wmma_bf16(a, bu, cu);
    }
#pragma unroll
    for (int j = 0; j < 8; ++j) {
      int m = hf ? (j + 8) : j;
      float h = fast_silu(cg[j]) * cu[j];             // silu(g)*u, v_rcp based
      hs[m * II + n0 + r] = (__bf16)h;
    }
  }
  __syncthreads();

  // ---- phase 2: out = h * Wd (scaled atomic-add for routed) ----
  for (int nt = wave; nt < NT2; nt += BLK / 32) {
    const int n0 = nt * 16;
    v8f c = {0, 0, 0, 0, 0, 0, 0, 0};
    for (int k0 = 0; k0 < II; k0 += 32) {
      __builtin_prefetch(wde + (size_t)(k0 + 32) * H + n0, 0, 0);
      v16bf a = lds_load_a(hs, II, k0);
      v16bf b = glb_load_b(wde + (size_t)k0 * H + n0, H);
      c = wmma_bf16(a, b, c);
    }
#pragma unroll
    for (int j = 0; j < 8; ++j) {
      int m = hf ? (j + 8) : j;
      int t = tok_s[m];
      if (ROUTED) {
        if (t >= 0) atomicAdd(out + (size_t)t * H + n0 + r, c[j] * scl_s[m]);
      } else {
        out[(size_t)t * H + n0 + r] = c[j];         // shared expert initializes out
      }
    }
  }
}

// ---------------- launch ----------------
extern "C" void kernel_launch(void* const* d_in, const int* in_sizes, int n_in,
                              void* d_out, int out_size, void* d_ws, size_t ws_size,
                              hipStream_t stream) {
  const float* x       = (const float*)d_in[0];
  const float* gate_w  = (const float*)d_in[1];
  const float* e_bias  = (const float*)d_in[2];
  const float* w_gate  = (const float*)d_in[3];
  const float* w_up    = (const float*)d_in[4];
  const float* w_down  = (const float*)d_in[5];
  const float* sw_gate = (const float*)d_in[6];
  const float* sw_up   = (const float*)d_in[7];
  const float* sw_down = (const float*)d_in[8];
  float* out = (float*)d_out;

  char* ws = (char*)d_ws;
  float* topk_w  = (float*)ws;                           // T*TOPK floats
  int*   topk_id = (int*)(ws + T * TOPK * 4);            // T*TOPK ints
  int*   cnt     = (int*)(ws + 2 * T * TOPK * 4);        // E ints
  int*   slot_tok= (int*)(ws + 2 * T * TOPK * 4 + 128);  // E*CAP ints
  float* slot_w  = (float*)(ws + 2 * T * TOPK * 4 + 128 + E * CAP * 4); // E*CAP floats

  const size_t smem = (size_t)MT * H * 4               // fp32 TDM stage
                    + (size_t)(MT * H + MT * II) * 2   // bf16 x + h tiles
                    + MT * 8;                          // tok + scale

  router_kernel<<<T, 32, 0, stream>>>(x, gate_w, e_bias, topk_w, topk_id);
  zero_counts<<<1, 32, 0, stream>>>(cnt);
  scatter_kernel<<<(T + 255) / 256, 256, 0, stream>>>(topk_id, topk_w, cnt, slot_tok, slot_w);
  // shared expert first: plain stores initialize d_out
  ffn_kernel<false><<<T / MT, BLK, smem, stream>>>(
      x, sw_gate, sw_up, sw_down, nullptr, nullptr, nullptr, out);
  // routed experts: expert-major grid for L2 weight reuse; early-exit on real tile count
  ffn_kernel<true><<<dim3(CAP / MT, E), BLK, smem, stream>>>(
      x, w_gate, w_up, w_down, slot_tok, slot_w, cnt, out);
}
